// Cross_Attention_73254962200935
// MI455X (gfx1250) — compile-verified, matrix operand order checked by reading, not run on hardware
//
#include <hip/hip_runtime.h>
#include <hip/hip_bf16.h>
#include <math.h>

// ---------------------------------------------------------------------------
// Cross-attention for MI455X (gfx1250, wave32, WMMA).
// b=1, n=2048, dim=1024, heads=16, d_head=64.
// Pipeline: f32->f16 convert/transpose -> WMMA QKV GEMMs (64x64 reg tile,
// V stored transposed) -> flash attention (32 queries/wave, online softmax,
// WMMA f16) -> WMMA out GEMM (64x64 reg tile).
// ---------------------------------------------------------------------------

typedef __attribute__((ext_vector_type(16))) _Float16 v16h;
typedef __attribute__((ext_vector_type(8)))  _Float16 v8h;
typedef __attribute__((ext_vector_type(8)))  float    v8f;

#define N_SEQ 2048
#define DIM   1024
#define HEADS 16
#define DHEAD 64
#define SCALE 0.03125f            // dim^-0.5 = 1/32
#define NEGMAX (-3.4028234663852886e38f)

// Load a 16x32 f16 WMMA fragment row for this lane.
// Per ISA 7.12.2 (16-bit A, wave32): lane L holds row M=L%16;
// lanes 0-15: K = {0..7, 16..23}; lanes 16-31: K = {8..15, 24..31}.
// Caller passes p = row_base + k_base + ((lane>>4)<<3); we read p[0..7], p[16..23].
static __device__ __forceinline__ v16h load_frag16(const _Float16* p) {
  union { v16h v; struct { v8h lo; v8h hi; } s; } u;
  u.s.lo = *(const v8h*)(p);
  u.s.hi = *(const v8h*)(p + 16);
  return u.v;
}

static __device__ __forceinline__ v8f wmma_f16(v16h a, v16h b, v8f c) {
  return __builtin_amdgcn_wmma_f32_16x16x32_f16(false, a, false, b,
                                                (short)0, c, false, false);
}

// ---------------------------------------------------------------------------
// Elementwise f32 -> f16
// ---------------------------------------------------------------------------
__global__ void cvt_f16_kernel(const float* __restrict__ src,
                               _Float16* __restrict__ dst, int n) {
  int i = blockIdx.x * blockDim.x + threadIdx.x;
  if (i < n) dst[i] = (_Float16)src[i];
}

// Transposing f32 [K][Nc] -> f16 [Nc][K] (coalesced writes)
__global__ void transpose_f16_kernel(const float* __restrict__ src,
                                     _Float16* __restrict__ dst, int K, int Nc) {
  int i = blockIdx.x * blockDim.x + threadIdx.x;
  if (i < K * Nc) {
    int nrow = i / K;
    int kc   = i - nrow * K;
    dst[i] = (_Float16)src[(size_t)kc * Nc + nrow];
  }
}

// ---------------------------------------------------------------------------
// Projection GEMM: C[2048,1024] = A[2048,1024] @ B, with Bt = B^T (f16).
// 64x64 output tile per wave: 4 A-frags x 4 B-frags -> 16 WMMAs per k-step.
// transpose_out==0: store f16 head-major  out[h][row][d]  (Q, K)
// transpose_out==1: store f16 transposed  out[h][d][row]  (V^T for PV WMMA)
// ---------------------------------------------------------------------------
__global__ __launch_bounds__(256) void gemm_qkv_kernel(
    const _Float16* __restrict__ A, const _Float16* __restrict__ Bt,
    _Float16* __restrict__ out, int transpose_out) {
  const int wid  = threadIdx.x >> 5;
  const int lane = threadIdx.x & 31;
  const int gid  = blockIdx.x * 8 + wid;
  const int tm   = gid >> 4;          // 16 column tiles (N=1024/64)
  const int tn   = gid & 15;
  const int m    = lane & 15;
  const int off  = (lane >> 4) << 3;
  const int row8 = (lane >> 4) << 3;

  const _Float16* a0 = A  + (size_t)(tm * 64 + m) * DIM + off;
  const _Float16* b0 = Bt + (size_t)(tn * 64 + m) * DIM + off;

  v8f acc[4][4] = {};
  for (int kk = 0; kk < DIM; kk += 32) {
    v16h af[4], bf[4];
#pragma unroll
    for (int i = 0; i < 4; ++i) {
      af[i] = load_frag16(a0 + (size_t)(i * 16) * DIM + kk);
      bf[i] = load_frag16(b0 + (size_t)(i * 16) * DIM + kk);
    }
#pragma unroll
    for (int mi = 0; mi < 4; ++mi)
#pragma unroll
      for (int ni = 0; ni < 4; ++ni)
        acc[mi][ni] = wmma_f16(af[mi], bf[ni], acc[mi][ni]);
  }

  // C/D layout: element r of lane L -> row = r + 8*(L>=16), col = L%16
#pragma unroll
  for (int mi = 0; mi < 4; ++mi) {
#pragma unroll
    for (int ni = 0; ni < 4; ++ni) {
      const int col = tn * 64 + ni * 16 + m;
      const int h   = col >> 6;
      const int d   = col & 63;
#pragma unroll
      for (int r = 0; r < 8; ++r) {
        const int row = tm * 64 + mi * 16 + row8 + r;
        const size_t idx = transpose_out
            ? ((size_t)(h * DHEAD + d) * N_SEQ + row)   // V^T: [h][d][n]
            : ((size_t)(h * N_SEQ + row) * DHEAD + d);  // Q/K: [h][n][d]
        out[idx] = (_Float16)acc[mi][ni][r];
      }
    }
  }
}

// ---------------------------------------------------------------------------
// Flash attention: one wave = one (head, 32-query block).
// Iterate 2048 keys in 32-key blocks: S = Q K^T (8 WMMA), online softmax in
// LDS (each of the 32 lanes owns one row), rescale accs, O += P V (8 WMMA,
// V fragments shared across both query sub-tiles).
// ---------------------------------------------------------------------------
__global__ __launch_bounds__(256) void attn_kernel(
    const _Float16* __restrict__ Q, const _Float16* __restrict__ K,
    const _Float16* __restrict__ Vt, const unsigned char* __restrict__ mask,
    _Float16* __restrict__ O) {
  __shared__ __align__(16) float    Sb[8][32][32];
  __shared__ __align__(16) _Float16 Pb[8][32][32];
  __shared__ __align__(16) float    Cb[8][32];

  const int wid  = threadIdx.x >> 5;
  const int lane = threadIdx.x & 31;
  const int gid  = blockIdx.x * 8 + wid;
  const int h    = gid >> 6;              // 64 query blocks (of 32) per head
  const int qb   = (gid & 63) << 5;

  const _Float16* Qh = Q  + (size_t)h * N_SEQ * DHEAD;
  const _Float16* Kh = K  + (size_t)h * N_SEQ * DHEAD;
  const _Float16* Vh = Vt + (size_t)h * DHEAD * N_SEQ;

  const int m    = lane & 15;
  const int off  = (lane >> 4) << 3;
  const int row8 = (lane >> 4) << 3;

  // Q fragments (2 sub-tiles of 16 rows x 64 k), loaded once.
  v16h qa[2][2];
#pragma unroll
  for (int mi = 0; mi < 2; ++mi)
#pragma unroll
    for (int kt = 0; kt < 2; ++kt)
      qa[mi][kt] = load_frag16(Qh + (size_t)(qb + mi * 16 + m) * DHEAD +
                               kt * 32 + off);

  v8f o[2][4] = {};
  // Per-row running softmax state: lane L owns row (qb + L).
  float rm = NEGMAX, rl = 0.0f;
  const int irow = qb + lane;
  const bool pmI = (irow == 0) || (mask[irow - 1] != 0);

  for (int jb = 0; jb < N_SEQ; jb += 32) {
    // ---- S = Q K^T for 32x32 block (2x2 tiles of 16x16) -> LDS
#pragma unroll
    for (int jt = 0; jt < 2; ++jt) {
      const int nk = jb + (jt << 4) + m;   // key row index for B fragment
      const v16h kb0 = load_frag16(Kh + (size_t)nk * DHEAD + 0  + off);
      const v16h kb1 = load_frag16(Kh + (size_t)nk * DHEAD + 32 + off);
#pragma unroll
      for (int mi = 0; mi < 2; ++mi) {
        v8f s = {};
        s = wmma_f16(qa[mi][0], kb0, s);
        s = wmma_f16(qa[mi][1], kb1, s);
#pragma unroll
        for (int r = 0; r < 8; ++r)
          Sb[wid][mi * 16 + row8 + r][(jt << 4) + m] = s[r];
      }
    }
    asm volatile("s_wait_dscnt 0x0" ::: "memory");  // same-wave LDS RAW

    // ---- online softmax: lane L owns row L (all 32 lanes active)
    {
      float bm = NEGMAX;
#pragma unroll
      for (int j = 0; j < 32; ++j) {
        float s = Sb[wid][lane][j] * SCALE;
        const int jj = jb + j;
        const bool pmj = (jj == 0) || (mask[jj - 1] != 0);
        if (!(pmI && pmj)) s = NEGMAX;
        Sb[wid][lane][j] = s;
        bm = fmaxf(bm, s);
      }
      const float mnew = fmaxf(rm, bm);
      const float c    = __expf(rm - mnew);
      float sum = 0.0f;
#pragma unroll
      for (int j = 0; j < 32; ++j) {
        const float e = __expf(Sb[wid][lane][j] - mnew);
        sum += e;
        Pb[wid][lane][j] = (_Float16)e;
      }
      rl = rl * c + sum;
      rm = mnew;
      Cb[wid][lane] = c;
    }
    asm volatile("s_wait_dscnt 0x0" ::: "memory");

    // ---- rescale accumulators by per-row correction factor
#pragma unroll
    for (int mi = 0; mi < 2; ++mi)
#pragma unroll
      for (int r = 0; r < 8; ++r) {
        const float c = Cb[wid][mi * 16 + row8 + r];
#pragma unroll
        for (int t = 0; t < 4; ++t) o[mi][t][r] *= c;
      }

    // ---- O += P @ V  (P from LDS; V^T rows contiguous along n;
    //      V fragments shared by both query sub-tiles)
    v16h pa[2];
#pragma unroll
    for (int mi = 0; mi < 2; ++mi)
      pa[mi] = load_frag16(&Pb[wid][mi * 16 + m][off]);
#pragma unroll
    for (int t = 0; t < 4; ++t) {
      const v16h vb =
          load_frag16(Vh + (size_t)(t * 16 + m) * N_SEQ + jb + off);
#pragma unroll
      for (int mi = 0; mi < 2; ++mi)
        o[mi][t] = wmma_f16(pa[mi], vb, o[mi][t]);
    }
  }

  // ---- epilogue: divide by running sum, write f16 [n][h*64+d]
  Cb[wid][lane] = 1.0f / rl;
  asm volatile("s_wait_dscnt 0x0" ::: "memory");
#pragma unroll
  for (int mi = 0; mi < 2; ++mi) {
#pragma unroll
    for (int r = 0; r < 8; ++r) {
      const float inv = Cb[wid][mi * 16 + row8 + r];
      const int row = qb + mi * 16 + row8 + r;
      _Float16* orow = O + (size_t)row * DIM + h * DHEAD + m;
      orow[0]  = (_Float16)(o[mi][0][r] * inv);
      orow[16] = (_Float16)(o[mi][1][r] * inv);
      orow[32] = (_Float16)(o[mi][2][r] * inv);
      orow[48] = (_Float16)(o[mi][3][r] * inv);
    }
  }
}

// ---------------------------------------------------------------------------
// Output projection: out_f32[2048,1024] = O16 @ Wo + bo  (64x64 reg tile)
// ---------------------------------------------------------------------------
__global__ __launch_bounds__(256) void gemm_out_kernel(
    const _Float16* __restrict__ A, const _Float16* __restrict__ Bt,
    const float* __restrict__ bias, float* __restrict__ out) {
  const int wid  = threadIdx.x >> 5;
  const int lane = threadIdx.x & 31;
  const int gid  = blockIdx.x * 8 + wid;
  const int tm   = gid >> 4;
  const int tn   = gid & 15;
  const int m    = lane & 15;
  const int off  = (lane >> 4) << 3;
  const int row8 = (lane >> 4) << 3;

  const _Float16* a0 = A  + (size_t)(tm * 64 + m) * DIM + off;
  const _Float16* b0 = Bt + (size_t)(tn * 64 + m) * DIM + off;

  v8f acc[4][4] = {};
  for (int kk = 0; kk < DIM; kk += 32) {
    v16h af[4], bf[4];
#pragma unroll
    for (int i = 0; i < 4; ++i) {
      af[i] = load_frag16(a0 + (size_t)(i * 16) * DIM + kk);
      bf[i] = load_frag16(b0 + (size_t)(i * 16) * DIM + kk);
    }
#pragma unroll
    for (int mi = 0; mi < 4; ++mi)
#pragma unroll
      for (int ni = 0; ni < 4; ++ni)
        acc[mi][ni] = wmma_f16(af[mi], bf[ni], acc[mi][ni]);
  }

#pragma unroll
  for (int mi = 0; mi < 4; ++mi) {
#pragma unroll
    for (int ni = 0; ni < 4; ++ni) {
      const int col = tn * 64 + ni * 16 + m;
      const float b = bias[col];
#pragma unroll
      for (int r = 0; r < 8; ++r)
        out[(size_t)(tm * 64 + mi * 16 + row8 + r) * DIM + col] =
            acc[mi][ni][r] + b;
    }
  }
}

// ---------------------------------------------------------------------------
extern "C" void kernel_launch(void* const* d_in, const int* in_sizes, int n_in,
                              void* d_out, int out_size, void* d_ws, size_t ws_size,
                              hipStream_t stream) {
  (void)in_sizes; (void)n_in; (void)out_size; (void)ws_size;
  const float* x   = (const float*)d_in[0];
  const float* mem = (const float*)d_in[1];
  const unsigned char* mask = (const unsigned char*)d_in[2]; // bool mask, n-1
  const float* Wq  = (const float*)d_in[3];
  const float* Wk  = (const float*)d_in[4];
  const float* Wv  = (const float*)d_in[5];
  const float* Wo  = (const float*)d_in[6];
  const float* bo  = (const float*)d_in[7];
  float* out = (float*)d_out;

  // f16 workspace carve-up (~32 MB total)
  const size_t SZ_X = (size_t)N_SEQ * DIM;  // activation-sized (halves)
  const size_t SZ_W = (size_t)DIM * DIM;    // weight-sized (halves)
  _Float16* ws   = (_Float16*)d_ws;
  _Float16* x16  = ws;
  _Float16* m16  = x16  + SZ_X;
  _Float16* WqT  = m16  + SZ_X;
  _Float16* WkT  = WqT  + SZ_W;
  _Float16* WvT  = WkT  + SZ_W;
  _Float16* WoT  = WvT  + SZ_W;
  _Float16* Q16  = WoT  + SZ_W;   // [h][n][d]
  _Float16* K16  = Q16  + SZ_X;   // [h][n][d]
  _Float16* Vt16 = K16  + SZ_X;   // [h][d][n]
  _Float16* O16  = Vt16 + SZ_X;   // [n][h*d]

  const int nx = N_SEQ * DIM;
  const int nw = DIM * DIM;
  cvt_f16_kernel<<<(nx + 255) / 256, 256, 0, stream>>>(x,   x16, nx);
  cvt_f16_kernel<<<(nx + 255) / 256, 256, 0, stream>>>(mem, m16, nx);
  transpose_f16_kernel<<<(nw + 255) / 256, 256, 0, stream>>>(Wq, WqT, DIM, DIM);
  transpose_f16_kernel<<<(nw + 255) / 256, 256, 0, stream>>>(Wk, WkT, DIM, DIM);
  transpose_f16_kernel<<<(nw + 255) / 256, 256, 0, stream>>>(Wv, WvT, DIM, DIM);
  transpose_f16_kernel<<<(nw + 255) / 256, 256, 0, stream>>>(Wo, WoT, DIM, DIM);

  // 64x64 tiles: (2048/64)*(1024/64) = 512 waves -> 64 blocks x 8 waves
  const int gemm_blocks = (N_SEQ / 64) * (DIM / 64) / 8;
  gemm_qkv_kernel<<<gemm_blocks, 256, 0, stream>>>(x16, WqT, Q16,  0);
  gemm_qkv_kernel<<<gemm_blocks, 256, 0, stream>>>(m16, WkT, K16,  0);
  gemm_qkv_kernel<<<gemm_blocks, 256, 0, stream>>>(m16, WvT, Vt16, 1);

  // 32 queries per wave: 16 heads * 64 blocks = 1024 waves -> 128 blocks
  attn_kernel<<<(HEADS * (N_SEQ / 32)) / 8, 256, 0, stream>>>(Q16, K16, Vt16,
                                                              mask, O16);

  gemm_out_kernel<<<gemm_blocks, 256, 0, stream>>>(O16, WoT, bo, out);
}
